// MultiHeadAttention_84267258347924
// MI455X (gfx1250) — compile-verified, matrix operand order checked by reading, not run on hardware
//
#include <hip/hip_runtime.h>
#include <hip/hip_bf16.h>

typedef __attribute__((ext_vector_type(16))) _Float16 v16h;
typedef __attribute__((ext_vector_type(8)))  _Float16 v8h;
typedef __attribute__((ext_vector_type(8)))  float    v8f;

#define BB 4
#define SS 2048
#define DD 1024
#define HH 16
#define HD 64

// ---------------------------------------------------------------------------
// helpers
// ---------------------------------------------------------------------------
// Uniform base + 32-bit lane offset -> SADDR-form global_load_b128 pairs
__device__ __forceinline__ v16h load_frag(const _Float16* __restrict__ base, int off) {
    v8h lo = *(const v8h*)(base + off);
    v8h hi = *(const v8h*)(base + off + 16);
    v16h r;
#pragma unroll
    for (int i = 0; i < 8; ++i) { r[i] = lo[i]; r[8 + i] = hi[i]; }
    return r;
}

__device__ __forceinline__ v8f wmma_f16(v16h a, v16h b, v8f c) {
    // D = A(16x32 f16) x B(32x16 f16) + C(16x16 f32)
    return __builtin_amdgcn_wmma_f32_16x16x32_f16(false, a, false, b,
                                                  (short)0, c, false, false);
}

// ---------------------------------------------------------------------------
// casts / transposes (fp32 -> f16)
// ---------------------------------------------------------------------------
__global__ void cast_f32_f16(const float* __restrict__ in,
                             _Float16* __restrict__ out, int n) {
    int i = blockIdx.x * blockDim.x + threadIdx.x;
    if (i < n) out[i] = (_Float16)in[i];
}

// in: rows x cols (row-major) -> out: cols x rows (row-major), casting to f16
__global__ void transpose_cast(const float* __restrict__ in,
                               _Float16* __restrict__ out, int rows, int cols) {
    int i = blockIdx.x * blockDim.x + threadIdx.x;
    if (i < rows * cols) {
        int r = i / cols, c = i % cols;
        out[(size_t)c * rows + r] = (_Float16)in[i];
    }
}

// ---------------------------------------------------------------------------
// QKV projection, 64x64 wave tile (4x4 WMMA register blocking):
// qkv = x16 @ Wqkv + b, scattered into q/k (B,H,S,HD) and v^T (B,H,HD,S).
// Per k-step: 16 b128 loads feed 16 WMMAs (1:1).
// ---------------------------------------------------------------------------
__global__ void __launch_bounds__(256, 2)
qkv_gemm(const _Float16* __restrict__ x16,
         const _Float16* __restrict__ wT,
         const float* __restrict__ bias,
         _Float16* __restrict__ q16,
         _Float16* __restrict__ k16,
         _Float16* __restrict__ vt16) {
    const int lane = threadIdx.x & 31;
    const int wave = threadIdx.x >> 5;
    const int tile = blockIdx.x * 8 + wave;          // (8192/64)*(3072/64)=6144
    const int nt = tile % 48, mt = tile / 48;
    const int hi = lane >> 4, ln = lane & 15, koff = hi * 8;

    // uniform bases (SGPRs); per-lane 32-bit offsets
    const _Float16* abase = x16 + (size_t)(mt * 64) * DD;
    const _Float16* bbase = wT  + (size_t)(nt * 64) * DD;
    int aoff[4], boff[4];
#pragma unroll
    for (int i = 0; i < 4; ++i) {
        aoff[i] = (i * 16 + ln) * DD + koff;
        boff[i] = (i * 16 + ln) * DD + koff;
    }

    v8f acc[16];
#pragma unroll
    for (int i = 0; i < 16; ++i) acc[i] = (v8f){};

    for (int kk = 0; kk < DD; kk += 32) {
        __builtin_prefetch(abase + aoff[0] + kk + 256, 0, 3);
        __builtin_prefetch(bbase + boff[0] + kk + 256, 0, 3);
        v16h a[4], b[4];
#pragma unroll
        for (int i = 0; i < 4; ++i) a[i] = load_frag(abase, aoff[i] + kk);
#pragma unroll
        for (int j = 0; j < 4; ++j) b[j] = load_frag(bbase, boff[j] + kk);
#pragma unroll
        for (int i = 0; i < 4; ++i)
#pragma unroll
            for (int j = 0; j < 4; ++j)
                acc[i * 4 + j] = wmma_f16(a[i], b[j], acc[i * 4 + j]);
    }

#pragma unroll
    for (int j = 0; j < 4; ++j) {
        const int col = nt * 64 + j * 16 + ln;       // 0..3071
        const float bv = bias[col];
        const int h = col / 192, cc = col % 192;     // head-major [q|k|v] blocks
#pragma unroll
        for (int i = 0; i < 4; ++i) {
#pragma unroll
            for (int r = 0; r < 8; ++r) {
                int row = mt * 64 + i * 16 + 8 * hi + r;   // 0..8191
                int bidx = row >> 11, s = row & (SS - 1);
                _Float16 hv = (_Float16)(acc[i * 4 + j][r] + bv);
                int head = bidx * HH + h;
                if (cc < 64)       q16[(head * SS + s) * HD + cc]          = hv;
                else if (cc < 128) k16[(head * SS + s) * HD + (cc - 64)]   = hv;
                else               vt16[(head * HD + (cc - 128)) * SS + s] = hv;
            }
        }
    }
}

// ---------------------------------------------------------------------------
// Flash attention: one wave per (b,h, 32 query rows = two 16-row q-tiles).
// S^T = K Q^T so softmax over keys is lane-local + xor16; the f32 D-layout of
// S^T maps lane-locally onto the P A-fragment for P@V. K/V fragment loads are
// shared across both q-tiles: 16 WMMAs per 32-key step vs 16 b128 loads.
// ---------------------------------------------------------------------------
__global__ void __launch_bounds__(256, 2)
flash_attn(const _Float16* __restrict__ q16,
           const _Float16* __restrict__ k16,
           const _Float16* __restrict__ vt16,
           _Float16* __restrict__ attn16) {
    const int lane = threadIdx.x & 31;
    const int wave = threadIdx.x >> 5;
    const int tile = blockIdx.x * 8 + wave;          // 0..4095
    const int qt = tile & (SS / 32 - 1);             // 64 q-blocks per head
    const int bh = tile >> 6;                        // 0..63
    const int qbase = qt * 32;
    const int hi = lane >> 4, ln = lane & 15, koff = hi * 8;
    const float scale = 0.125f;                      // 1/sqrt(64)

    const _Float16* qb = q16 + (size_t)bh * SS * HD;
    const _Float16* kb = k16 + (size_t)bh * SS * HD;
    const _Float16* vb = vt16 + (size_t)bh * HD * SS;

    // Q^T B-fragments for both q-tiles (hd is the WMMA K dim), loaded once
    v16h Bq0[2], Bq1[2];
#pragma unroll
    for (int u = 0; u < 2; ++u) {
        int qo = (qbase + u * 16 + ln) * HD;
        Bq0[u] = load_frag(qb, qo + koff);
        Bq1[u] = load_frag(qb, qo + 32 + koff);
    }

    float m[2] = {-1e30f, -1e30f}, l[2] = {0.0f, 0.0f};
    v8f C[2][4];
#pragma unroll
    for (int u = 0; u < 2; ++u)
#pragma unroll
        for (int c = 0; c < 4; ++c) C[u][c] = (v8f){};

    for (int j = 0; j < SS; j += 32) {
        // ---- K-tile A-fragments (shared by both q-tiles)
        const int ko = (j + ln) * HD + koff;
        __builtin_prefetch(kb + ko + 32 * HD, 0, 3);
        v16h A0a = load_frag(kb, ko);
        v16h A0b = load_frag(kb, ko + 32);
        v16h A1a = load_frag(kb, ko + 16 * HD);
        v16h A1b = load_frag(kb, ko + 16 * HD + 32);

        v16h pa[2];
#pragma unroll
        for (int u = 0; u < 2; ++u) {
            // scores^T, two 16-key tiles for this q-tile
            v8f s0 = {}, s1 = {};
            s0 = wmma_f16(A0a, Bq0[u], s0);
            s0 = wmma_f16(A0b, Bq1[u], s0);
            s1 = wmma_f16(A1a, Bq0[u], s1);
            s1 = wmma_f16(A1b, Bq1[u], s1);

            // online softmax (per query column; lane + xor16 partner)
            float mx = -1e30f;
#pragma unroll
            for (int r = 0; r < 8; ++r) mx = fmaxf(mx, fmaxf(s0[r], s1[r]));
            mx *= scale;
            mx = fmaxf(mx, __shfl_xor(mx, 16, 32));
            float mn = fmaxf(m[u], mx);
            float alpha = __expf(m[u] - mn);
            m[u] = mn;

            float p0, p1, lsum = 0.0f;
#pragma unroll
            for (int r = 0; r < 8; ++r) {
                p0 = __expf(s0[r] * scale - mn);
                p1 = __expf(s1[r] * scale - mn);
                lsum += p0 + p1;
                pa[u][r]     = (_Float16)p0;   // D-layout of S^T maps
                pa[u][8 + r] = (_Float16)p1;   // lane-locally onto P A-frag
            }
            lsum += __shfl_xor(lsum, 16, 32);
            l[u] = l[u] * alpha + lsum;

            // rescale accumulators: per-row factor lives in lane (8*hi + r)
#pragma unroll
            for (int r = 0; r < 8; ++r) {
                float f = __shfl(alpha, koff + r, 32);
                C[u][0][r] *= f; C[u][1][r] *= f;
                C[u][2][r] *= f; C[u][3][r] *= f;
            }
        }

        // ---- P @ V: stream 4 V B-fragments, each feeds both q-tiles
#pragma unroll
        for (int c = 0; c < 4; ++c) {
            v16h Bv = load_frag(vb, (c * 16 + ln) * SS + j + koff);
            C[0][c] = wmma_f16(pa[0], Bv, C[0][c]);
            C[1][c] = wmma_f16(pa[1], Bv, C[1][c]);
        }
    }

    // ---- finalize: divide by l, write f16 (B,S,D) row-major
    const int bidx = bh >> 4, h = bh & 15;
#pragma unroll
    for (int u = 0; u < 2; ++u) {
#pragma unroll
        for (int r = 0; r < 8; ++r) {
            float lr = __shfl(l[u], koff + r, 32);
            float inv = 1.0f / lr;
            int s = qbase + u * 16 + koff + r;
            _Float16* op = attn16 + ((size_t)(bidx * SS + s)) * DD + h * HD + ln;
            op[0]  = (_Float16)(C[u][0][r] * inv);
            op[16] = (_Float16)(C[u][1][r] * inv);
            op[32] = (_Float16)(C[u][2][r] * inv);
            op[48] = (_Float16)(C[u][3][r] * inv);
        }
    }
}

// ---------------------------------------------------------------------------
// Output projection, 64x64 wave tile: out = attn16 @ Wout + b_out (fp32)
// ---------------------------------------------------------------------------
__global__ void __launch_bounds__(256, 2)
out_gemm(const _Float16* __restrict__ a16,
         const _Float16* __restrict__ wT,
         const float* __restrict__ bias,
         float* __restrict__ out) {
    const int lane = threadIdx.x & 31;
    const int wave = threadIdx.x >> 5;
    const int tile = blockIdx.x * 8 + wave;          // (8192/64)*(1024/64)=2048
    const int nt = tile % 16, mt = tile / 16;
    const int hi = lane >> 4, ln = lane & 15, koff = hi * 8;

    const _Float16* abase = a16 + (size_t)(mt * 64) * DD;
    const _Float16* bbase = wT  + (size_t)(nt * 64) * DD;
    int aoff[4], boff[4];
#pragma unroll
    for (int i = 0; i < 4; ++i) {
        aoff[i] = (i * 16 + ln) * DD + koff;
        boff[i] = (i * 16 + ln) * DD + koff;
    }

    v8f acc[16];
#pragma unroll
    for (int i = 0; i < 16; ++i) acc[i] = (v8f){};

    for (int kk = 0; kk < DD; kk += 32) {
        __builtin_prefetch(abase + aoff[0] + kk + 256, 0, 3);
        __builtin_prefetch(bbase + boff[0] + kk + 256, 0, 3);
        v16h a[4], b[4];
#pragma unroll
        for (int i = 0; i < 4; ++i) a[i] = load_frag(abase, aoff[i] + kk);
#pragma unroll
        for (int j = 0; j < 4; ++j) b[j] = load_frag(bbase, boff[j] + kk);
#pragma unroll
        for (int i = 0; i < 4; ++i)
#pragma unroll
            for (int j = 0; j < 4; ++j)
                acc[i * 4 + j] = wmma_f16(a[i], b[j], acc[i * 4 + j]);
    }

#pragma unroll
    for (int j = 0; j < 4; ++j) {
        const int col = nt * 64 + j * 16 + ln;
        const float bv = bias[col];
#pragma unroll
        for (int i = 0; i < 4; ++i) {
#pragma unroll
            for (int r = 0; r < 8; ++r) {
                int row = mt * 64 + i * 16 + 8 * hi + r;
                out[(size_t)row * DD + col] = acc[i * 4 + j][r] + bv;
            }
        }
    }
}

// ---------------------------------------------------------------------------
// host launcher
// ---------------------------------------------------------------------------
extern "C" void kernel_launch(void* const* d_in, const int* in_sizes, int n_in,
                              void* d_out, int out_size, void* d_ws, size_t ws_size,
                              hipStream_t stream) {
    const float* x    = (const float*)d_in[0];   // (B,S,D)
    const float* Wqkv = (const float*)d_in[1];   // (D,3D)
    const float* bqkv = (const float*)d_in[2];   // (3D)
    const float* Wout = (const float*)d_in[3];   // (D,D)
    const float* bout = (const float*)d_in[4];   // (D)
    float* out = (float*)d_out;

    char* ws = (char*)d_ws;
    size_t o = 0;
    _Float16* x16    = (_Float16*)(ws + o); o += (size_t)BB * SS * DD * 2;      // 16MB
    _Float16* wqkvT  = (_Float16*)(ws + o); o += (size_t)3 * DD * DD * 2;       //  6MB
    _Float16* woutT  = (_Float16*)(ws + o); o += (size_t)DD * DD * 2;           //  2MB
    _Float16* q16    = (_Float16*)(ws + o); o += (size_t)BB * SS * DD * 2;      // 16MB
    _Float16* k16    = (_Float16*)(ws + o); o += (size_t)BB * SS * DD * 2;      // 16MB
    _Float16* vt16   = (_Float16*)(ws + o); o += (size_t)BB * SS * DD * 2;      // 16MB
    _Float16* attn16 = (_Float16*)(ws + o); o += (size_t)BB * SS * DD * 2;      // 16MB

    const int nx = BB * SS * DD;                       // 8388608
    cast_f32_f16<<<(nx + 255) / 256, 256, 0, stream>>>(x, x16, nx);

    const int nwq = DD * 3 * DD;                       // 3145728
    transpose_cast<<<(nwq + 255) / 256, 256, 0, stream>>>(Wqkv, wqkvT, DD, 3 * DD);

    const int nwo = DD * DD;                           // 1048576
    transpose_cast<<<(nwo + 255) / 256, 256, 0, stream>>>(Wout, woutT, DD, DD);

    // QKV GEMM: (8192/64)*(3072/64) = 6144 wave-tiles, 8 waves/block
    qkv_gemm<<<6144 / 8, 256, 0, stream>>>(x16, wqkvT, bqkv, q16, k16, vt16);

    // Flash attention: B*H*(S/32) = 4096 waves, 8 waves/block
    flash_attn<<<4096 / 8, 256, 0, stream>>>(q16, k16, vt16, attn16);

    // Output GEMM: (8192/64)*(1024/64) = 2048 wave-tiles, 8 waves/block
    out_gemm<<<2048 / 8, 256, 0, stream>>>(attn16, woutT, bout, out);
}
